// LSTM_38482906972892
// MI455X (gfx1250) — matrix operand order repaired and checked by reading, not compiled, not run on hardware
//
#include <hip/hip_runtime.h>
#include <hip/hip_bf16.h>

// LSTM  T=2048, B=32, E=512, H=512
// Phase 1: x_proj[T*B,4H] = embeds[T*B,E] @ concat(Wix,Wfx,Wgx,Wox) + b   (WMMA bf16)
// Phase 2: 16 persistent WGs, Wh slice (512x128 bf16 = 128KB) resident in LDS,
//          one device-wide barrier per timestep, h ping-pong in global scratch,
//          per-step h staged via GLOBAL_LOAD_ASYNC_TO_LDS_B128 (ASYNCcnt).
//
// Workspace layout (assumes ws_size >= 512MiB + 64KiB + 256B):
//   [0)                x_proj fp32 [65536][2048]          = 536870912 B
//   [XP_BYTES)         h ping-pong bf16 [2][32][512]      = 65536 B
//   [XP_BYTES+HP)      barrier {count, generation} u32[2]

#define Tn 2048
#define Bn 32
#define En 512
#define Hn 512
#define N4H 2048
#define MROWS (Tn * Bn)          // 65536
#define XP_BYTES (536870912ull)  // MROWS * N4H * 4
#define HP_BYTES (65536ull)      // 2 * 32 * 512 * 2
#define NWG2 16                  // phase-2 workgroups

typedef __bf16 bf16;
typedef __attribute__((ext_vector_type(16))) __bf16 v16bf;
typedef __attribute__((ext_vector_type(8)))  __bf16 bf16x8;
typedef __attribute__((ext_vector_type(4)))  __bf16 bf16x4;
typedef __attribute__((ext_vector_type(8)))  float  v8f;
typedef __attribute__((ext_vector_type(4)))  float  f32x4;

// Async-LDS builtin wants int4-typed pointers (per hipcc diagnostic).
typedef __attribute__((__vector_size__(16))) int v4i;
typedef __attribute__((address_space(1))) v4i gv4i;   // global int4
typedef __attribute__((address_space(3))) v4i lv4i;   // LDS int4

#if defined(__has_builtin)
#if __has_builtin(__builtin_amdgcn_global_load_async_to_lds_b128)
#define HAVE_ASYNC_LDS 1
#else
#define HAVE_ASYNC_LDS 0
#warning "CDNA5 probe: __builtin_amdgcn_global_load_async_to_lds_b128 NOT available; using ds fallback"
#endif
#if __has_builtin(__builtin_amdgcn_s_wait_asynccnt)
#define HAVE_WAIT_ASYNC 1
#else
#define HAVE_WAIT_ASYNC 0
#warning "CDNA5 probe: __builtin_amdgcn_s_wait_asynccnt NOT available; using inline asm"
#endif
#else
#define HAVE_ASYNC_LDS 0
#define HAVE_WAIT_ASYNC 0
#endif

__device__ __forceinline__ void wait_asynccnt0() {
#if HAVE_WAIT_ASYNC
  __builtin_amdgcn_s_wait_asynccnt(0);
#else
  asm volatile("s_wait_asynccnt 0" ::: "memory");
#endif
}

union FragU { v16bf v; bf16x8 h[2]; };

__device__ __forceinline__ v8f wmma_bf16(v16bf a, v16bf b, v8f c) {
  // (neg_a, A, neg_b, B, c_mod, C, reuse_a, reuse_b)
  return __builtin_amdgcn_wmma_f32_16x16x32_bf16(false, a, false, b, (short)0, c,
                                                 false, false);
}

// A fragment: 16x32 tile, rows r0..r0+15, k chunk k0..k0+31, row-major [.., stride]
// ISA 16-bit A layout: lane&15 = row; lanes>=16 take K {8..15,24..31}.
__device__ __forceinline__ v16bf load_afrag(const bf16* p, int stride, int r0, int k0) {
  const int lane = threadIdx.x & 31;
  const int r  = r0 + (lane & 15);
  const int kh = (lane >> 4) * 8;
  FragU u;
  u.h[0] = *(const bf16x8*)(p + r * stride + k0 + kh);
  u.h[1] = *(const bf16x8*)(p + r * stride + k0 + 16 + kh);
  return u.v;
}

// B fragment: 32x16 tile from n-major storage p[n][k] (stride elems per n).
// lane&15 = column n; lanes 0-15 hold K {0..7,16..23}, lanes 16-31 hold K {8..15,24..31}.
__device__ __forceinline__ v16bf load_bfrag(const bf16* p, int stride, int n0, int k0) {
  const int lane = threadIdx.x & 31;
  const int n  = n0 + (lane & 15);
  const int kh = (lane >> 4) * 8;
  FragU u;
  u.h[0] = *(const bf16x8*)(p + n * stride + k0 + kh);
  u.h[1] = *(const bf16x8*)(p + n * stride + k0 + 16 + kh);
  return u.v;
}

// ---------------------------------------------------------------- init kernel
__global__ void lstm_init_kernel(bf16* __restrict__ hping, unsigned* __restrict__ bar) {
  for (int i = threadIdx.x; i < 2 * Bn * Hn; i += blockDim.x)
    hping[i] = (bf16)0.f;
  if (threadIdx.x == 0) { bar[0] = 0u; bar[1] = 0u; }
}

// ------------------------------------------------------- phase 1: x_proj GEMM
// Grid: (512, 16). Block tile 128(M) x 128(N), K steps of 32, 8 waves,
// wave tile 64x32 (acc[4][2] of v8f).
__launch_bounds__(256)
__global__ void xproj_kernel(const float* __restrict__ X,
                             const float* __restrict__ Wix, const float* __restrict__ Wfx,
                             const float* __restrict__ Wgx, const float* __restrict__ Wox,
                             const float* __restrict__ bi,  const float* __restrict__ bfv,
                             const float* __restrict__ bg,  const float* __restrict__ bo,
                             float* __restrict__ P) {
  __shared__ bf16 sA[128][40];   // [m][k], stride 40 (80B: 8B/16B aligned)
  __shared__ bf16 sB[128][40];   // [n][k]

  const int m0 = blockIdx.x * 128;
  const int n0 = blockIdx.y * 128;
  const int q  = n0 >> 9;            // gate block (N tile never crosses a gate)
  const int nc0 = n0 & 511;
  const float* Wq = (q == 0) ? Wix : (q == 1) ? Wfx : (q == 2) ? Wgx : Wox;
  const float* bq = (q == 0) ? bi  : (q == 1) ? bfv : (q == 2) ? bg  : bo;

  const int tid  = threadIdx.x;
  const int wave = tid >> 5, lane = tid & 31;
  const int m0w = (wave >> 2) * 64;  // 2 M groups of waves
  const int n0w = (wave & 3) * 32;   // 4 N groups of waves

  v8f acc[4][2];
#pragma unroll
  for (int mt = 0; mt < 4; ++mt)
#pragma unroll
    for (int nt = 0; nt < 2; ++nt) acc[mt][nt] = {};

  for (int k0 = 0; k0 < En; k0 += 32) {
    // stage A tile 128x32: float4 load (b128) -> packed bf16x4 store (b64)
#pragma unroll
    for (int i = 0; i < 4; ++i) {
      const int idx = tid + i * 256;           // 1024 float4 chunks
      const int r = idx >> 3, k4 = (idx & 7) * 4;
      const f32x4 f = *(const f32x4*)&X[(size_t)(m0 + r) * En + k0 + k4];
      bf16x4 pk = {(bf16)f[0], (bf16)f[1], (bf16)f[2], (bf16)f[3]};
      *(bf16x4*)&sA[r][k4] = pk;
    }
    // stage B tile 32x128 transposed to [n][k]: 4 coalesced loads along k,
    // one packed bf16x4 store per thread-chunk
#pragma unroll
    for (int i = 0; i < 4; ++i) {
      const int idx = tid + i * 256;           // 1024 chunks
      const int n = idx & 127, kk = (idx >> 7) * 4;
      bf16x4 pk;
#pragma unroll
      for (int j = 0; j < 4; ++j)
        pk[j] = (bf16)Wq[(size_t)(k0 + kk + j) * Hn + nc0 + n];
      *(bf16x4*)&sB[n][kk] = pk;
    }
    __syncthreads();

    v16bf af[4], bf2[2];
#pragma unroll
    for (int mt = 0; mt < 4; ++mt) af[mt] = load_afrag(&sA[0][0], 40, m0w + mt * 16, 0);
#pragma unroll
    for (int nt = 0; nt < 2; ++nt) bf2[nt] = load_bfrag(&sB[0][0], 40, n0w + nt * 16, 0);
#pragma unroll
    for (int mt = 0; mt < 4; ++mt)
#pragma unroll
      for (int nt = 0; nt < 2; ++nt)
        acc[mt][nt] = wmma_bf16(af[mt], bf2[nt], acc[mt][nt]);
    __syncthreads();
  }

  // bias + store (D layout: N = lane&15, M = j + 8*(lane>=16))
  const int col16 = lane & 15;
  const int rsel  = (lane >> 4) * 8;
#pragma unroll
  for (int nt = 0; nt < 2; ++nt) {
    const int gcol = n0 + n0w + nt * 16 + col16;
    const float bb = bq[gcol & 511];
#pragma unroll
    for (int mt = 0; mt < 4; ++mt) {
#pragma unroll
      for (int j = 0; j < 8; ++j) {
        const int row = m0 + m0w + mt * 16 + rsel + j;
        P[(size_t)row * N4H + gcol] = acc[mt][nt][j] + bb;
      }
    }
  }
}

// ------------------------------------------------ device-wide barrier (16 WGs)
__device__ __forceinline__ void grid_barrier(unsigned* cnt, volatile unsigned* gen,
                                             unsigned nwg) {
  __syncthreads();
  if (threadIdx.x == 0) {
    __threadfence();                       // release: flush h/out stores
    const unsigned old_gen = *gen;
    const unsigned prev = atomicAdd(cnt, 1u);
    if (prev == nwg - 1u) {
      *cnt = 0u;
      __threadfence();
      atomicAdd((unsigned*)gen, 1u);
    } else {
      while (*gen == old_gen) { __builtin_amdgcn_s_sleep(1); }
    }
  }
  __syncthreads();
  __threadfence();                         // acquire: invalidate stale lines
}

// ---------------------------------------------- phase 2: sequential recurrence
// 16 WGs x 256 threads. WG g owns h columns [g*32, g*32+32) i.e. gate columns
// q*512 + g*32 + j. Dynamic LDS: Wh slice [128][520] bf16 + hbuf [32][520] bf16
// + gates [32][128] f32 + c [32][32] f32  = 186880 B.
#define WSTR 520
#define SW_OFF   0
#define HB_OFF   (128 * WSTR * 2)                       // 133120
#define SG_OFF   (HB_OFF + 32 * WSTR * 2)               // +33280
#define SC_OFF   (SG_OFF + 32 * 128 * 4)                // +16384
#define SMEM2_BYTES (SC_OFF + 32 * 32 * 4)              // 186880

__launch_bounds__(256)
__global__ void lstm_seq_kernel(const float* __restrict__ P,
                                const float* __restrict__ Wih, const float* __restrict__ Wfh,
                                const float* __restrict__ Wgh, const float* __restrict__ Woh,
                                bf16* __restrict__ hping,      // [2][32][512]
                                unsigned* __restrict__ bar,    // {cnt, gen}
                                float* __restrict__ out) {     // [T][32][512]
  extern __shared__ char smem[];
  bf16*  sW   = (bf16*)(smem + SW_OFF);   // [nloc][k] n-major, stride WSTR
  bf16*  hbuf = (bf16*)(smem + HB_OFF);   // [32][WSTR]
  float* sg   = (float*)(smem + SG_OFF);  // [32][128]
  float* sc   = (float*)(smem + SC_OFF);  // [32][32]

  const int g    = blockIdx.x;
  const int tid  = threadIdx.x;
  const int wave = tid >> 5, lane = tid & 31;

  // One-time: load/convert this WG's Wh slice into LDS (stays resident).
  // 4 coalesced loads along k per packed bf16x4 store.
  for (int i = tid; i < (128 * Hn) / 4; i += 256) {
    const int nloc = i & 127, k4 = (i >> 7) * 4;
    const int q = nloc >> 5, j = nloc & 31;
    const float* wsrc = (q == 0) ? Wih : (q == 1) ? Wfh : (q == 2) ? Wgh : Woh;
    bf16x4 pk;
#pragma unroll
    for (int jj = 0; jj < 4; ++jj)
      pk[jj] = (bf16)wsrc[(size_t)(k4 + jj) * Hn + g * 32 + j];
    *(bf16x4*)&sW[nloc * WSTR + k4] = pk;
  }
  for (int i = tid; i < 32 * 32; i += 256) sc[i] = 0.f;
  __syncthreads();

  const int mtile = wave & 1;     // M tile (rows 0-15 / 16-31)
  const int nbase = wave >> 1;    // N tiles nbase and nbase+4

  for (int t = 0; t < Tn; ++t) {
    // stage full h (bf16, 32KB) from ping buffer into LDS
    const bf16* hsrc = hping + (size_t)(t & 1) * (Bn * Hn);
#pragma unroll
    for (int i2 = 0; i2 < (Bn * Hn) / (8 * 256); ++i2) {    // 8 iters
      const int i = tid + i2 * 256;
      const int r = i >> 6, c8 = (i & 63) * 8;
#if HAVE_ASYNC_LDS
      __builtin_amdgcn_global_load_async_to_lds_b128(
          (gv4i*)&hsrc[r * Hn + c8],
          (lv4i*)&hbuf[r * WSTR + c8],
          0, 0);
#else
      *(bf16x8*)&hbuf[r * WSTR + c8] = *(const bf16x8*)&hsrc[r * Hn + c8];
#endif
    }
    if (t + 1 < Tn)   // warm L2 for next step's gate slice
      __builtin_prefetch(P + (size_t)(t + 1) * (Bn * N4H) + (size_t)(tid >> 5) * N4H
                           + g * 32, 0, 1);
#if HAVE_ASYNC_LDS
    wait_asynccnt0();
#endif
    __syncthreads();

    // gates_raw = h @ Wh  (16 tiles of 16x16 across 8 waves, 2 tiles each)
#pragma unroll
    for (int nn = 0; nn < 2; ++nn) {
      const int ntile = nbase + nn * 4;
      v8f acc = {};
#pragma unroll
      for (int kc = 0; kc < Hn; kc += 32) {
        v16bf a = load_afrag(hbuf, WSTR, mtile * 16, kc);
        v16bf b = load_bfrag(sW, WSTR, ntile * 16, kc);
        acc = wmma_bf16(a, b, acc);
      }
      const int col   = ntile * 16 + (lane & 15);
      const int rbase = mtile * 16 + (lane >> 4) * 8;
#pragma unroll
      for (int j = 0; j < 8; ++j) sg[(rbase + j) * 128 + col] = acc[j];
    }
    __syncthreads();

    // elementwise gates + state update; c lives in LDS, h -> out + ping buffer
    bf16* hdst = hping + (size_t)((t + 1) & 1) * (Bn * Hn);
    const int col = tid & 31;
    for (int rr = tid >> 5; rr < Bn; rr += 8) {
      const size_t pbase = (size_t)t * (Bn * N4H) + (size_t)rr * N4H + g * 32 + col;
      const float xi = P[pbase +    0] + sg[rr * 128 +  0 + col];
      const float xf = P[pbase +  512] + sg[rr * 128 + 32 + col];
      const float xg = P[pbase + 1024] + sg[rr * 128 + 64 + col];
      const float xo = P[pbase + 1536] + sg[rr * 128 + 96 + col];
      const float ii = 1.f / (1.f + __expf(-xi));
      const float ff = 1.f / (1.f + __expf(-xf));
      const float gg = tanhf(xg);
      const float oo = 1.f / (1.f + __expf(-xo));
      const float cN = ff * sc[rr * 32 + col] + ii * gg;
      sc[rr * 32 + col] = cN;
      const float hN = oo * tanhf(cN);
      out[(size_t)t * (Bn * Hn) + (size_t)rr * Hn + g * 32 + col] = hN;
      hdst[rr * Hn + g * 32 + col] = (bf16)hN;
    }

    grid_barrier(&bar[0], &bar[1], gridDim.x);
  }
}

// ----------------------------------------------------------------- host side
extern "C" void kernel_launch(void* const* d_in, const int* in_sizes, int n_in,
                              void* d_out, int out_size, void* d_ws, size_t ws_size,
                              hipStream_t stream) {
  const float* X   = (const float*)d_in[0];
  const float* Wix = (const float*)d_in[1];
  const float* Wih = (const float*)d_in[2];
  const float* bi  = (const float*)d_in[3];
  const float* Wfx = (const float*)d_in[4];
  const float* Wfh = (const float*)d_in[5];
  const float* bfv = (const float*)d_in[6];
  const float* Wgx = (const float*)d_in[7];
  const float* Wgh = (const float*)d_in[8];
  const float* bg  = (const float*)d_in[9];
  const float* Wox = (const float*)d_in[10];
  const float* Woh = (const float*)d_in[11];
  const float* bo  = (const float*)d_in[12];
  float* out = (float*)d_out;

  char* ws = (char*)d_ws;
  float*    xproj = (float*)ws;                       // 512 MiB
  bf16*     hping = (bf16*)(ws + XP_BYTES);           // 64 KiB
  unsigned* bar   = (unsigned*)(ws + XP_BYTES + HP_BYTES);

  lstm_init_kernel<<<1, 256, 0, stream>>>(hping, bar);

  dim3 g1(MROWS / 128, N4H / 128);
  xproj_kernel<<<g1, 256, 0, stream>>>(X, Wix, Wfx, Wgx, Wox, bi, bfv, bg, bo, xproj);

  lstm_seq_kernel<<<NWG2, 256, SMEM2_BYTES, stream>>>(xproj, Wih, Wfh, Wgh, Woh,
                                                      hping, bar, out);
}
